// SSDiMBlock_41781441856312
// MI455X (gfx1250) — matrix operand order repaired and checked by reading, not compiled, not run on hardware
//
#include <hip/hip_runtime.h>
#include <hip/hip_bf16.h>
#include <math.h>

#define BB 16
#define LL 1024
#define HH 768
#define DSZ 16
#define DII 768
#define DRR 48
#define KK 4
#define MLPH 3072
#define NPATH 4

typedef __bf16 bf16_t;
typedef __attribute__((ext_vector_type(16))) __bf16 bf16x16;
typedef __attribute__((ext_vector_type(8)))  __bf16 bf16x8;
typedef __attribute__((ext_vector_type(8)))  float  f32x8;
typedef __attribute__((ext_vector_type(4)))  unsigned int u32x4;
typedef __attribute__((ext_vector_type(8)))  int   i32x8;
typedef __attribute__((ext_vector_type(4)))  int   i32x4;

// ---------------- block reduction helper ----------------
__device__ __forceinline__ float blk_sum(float v) {
  __shared__ float red[256];
  int tid = threadIdx.x;
  red[tid] = v; __syncthreads();
  #pragma unroll
  for (int off = 128; off > 0; off >>= 1) {
    if (tid < off) red[tid] += red[tid + off];
    __syncthreads();
  }
  float r = red[0];
  __syncthreads();
  return r;
}

// ---------------- LN + modulate (optionally fused residual) ----------------
template<int MODE>
__global__ __launch_bounds__(256)
void k_ln_mod(const float* __restrict__ xin, const float* __restrict__ t,
              const float* __restrict__ mam, const float* __restrict__ yin,
              float* __restrict__ xout, bf16_t* __restrict__ hmod,
              int shiftSel, int scaleSel, int gateSel)
{
  const int token = blockIdx.x;
  const int b = token >> 10;               // / LL
  const size_t row = (size_t)token * HH;
  const float* tb = t + (size_t)b * 6 * HH;
  float v[3];
  float s = 0.f;
  #pragma unroll
  for (int i = 0; i < 3; i++) {
    int d = threadIdx.x + i * 256;
    float val = xin[row + d];
    if constexpr (MODE == 1) {
      val += tb[gateSel * HH + d] * mam[row + d] + yin[row + d];
      xout[row + d] = val;
    }
    v[i] = val; s += val;
  }
  float mean = blk_sum(s) * (1.0f / HH);
  float q = 0.f;
  #pragma unroll
  for (int i = 0; i < 3; i++) { float dl = v[i] - mean; q += dl * dl; }
  float var = blk_sum(q) * (1.0f / HH);
  float rstd = rsqrtf(var + 1e-6f);
  #pragma unroll
  for (int i = 0; i < 3; i++) {
    int d = threadIdx.x + i * 256;
    float nv = (v[i] - mean) * rstd;
    nv = nv * (1.0f + tb[scaleSel * HH + d]) + tb[shiftSel * HH + d];
    hmod[row + d] = (bf16_t)nv;
  }
}

// ---------------- WMMA GEMM: C[M,N] = A[M,K] * Bw[N,K]^T (+epilogue) ------
#define BM 128
#define BN 64
#define BK 32
#define LDSP 8
#define LDS_S (BK + LDSP)   // 40 halves per row (80B, conflict-free)

enum { EPI_NONE = 0, EPI_BIAS = 1, EPI_BIAS_GELU = 2, EPI_BIAS_SOFTPLUS = 3 };

// Tensor Data Mover: 2D bf16 tile (BK x BM) -> LDS with per-row padding so the
// LDS image matches the 80B-stride layout used by the fragment loads.
// D# per CDNA5 ISA ch.8: group0 = {flags, lds_addr, global_addr, type=2},
// group1 = {data_size=2B, pad_en, pad_interval=16dw, pad_amount=4dw, dims}.
__device__ __forceinline__ void tdm_load_a_bf16(
    void* lds_ptr, const void* gptr, int kRem, int mRem, int lda_elems)
{
  unsigned long long ga = (unsigned long long)(uintptr_t)gptr;
  unsigned lds_addr = (unsigned)(uintptr_t)lds_ptr;   // low 32 bits = LDS offset
  u32x4 g0;
  g0[0] = 1u;                                          // count=1, user descriptor
  g0[1] = lds_addr;                                    // lds_addr [63:32]
  g0[2] = (unsigned)(ga & 0xFFFFFFFFu);                // global_addr [95:64]
  g0[3] = (unsigned)((ga >> 32) & 0x1FFFFFFu) | (2u << 30);  // [120:96] | type=2
  unsigned td0 = (unsigned)kRem;                       // tensor_dim0 (K left, elems)
  unsigned td1 = (unsigned)mRem;                       // tensor_dim1 (rows left)
  i32x8 g1;
  g1[0] = (int)((1u << 16) | (1u << 20) | (3u << 22) | (3u << 25));
  g1[1] = (int)((td0 & 0xFFFFu) << 16);                // abar=0 | dim0.lo16
  g1[2] = (int)((td0 >> 16) | ((td1 & 0xFFFFu) << 16));// dim0.hi16 | dim1.lo16
  g1[3] = (int)((td1 >> 16) | ((unsigned)BK << 16));   // dim1.hi16 | tile_dim0=32
  g1[4] = (int)(unsigned)BM;                           // tile_dim1=128, tile_dim2=0
  g1[5] = (int)(unsigned)lda_elems;                    // tensor_dim0_stride lo32
  g1[6] = 0;                                           // stride0 hi16 | stride1 lo16
  g1[7] = 0;
  i32x4 z4 = {0, 0, 0, 0};
  i32x8 z8 = {0, 0, 0, 0, 0, 0, 0, 0};
  __builtin_amdgcn_tensor_load_to_lds(g0, g1, z4, z4, z8, 0);
}

template<int EPI, bool ABF, bool OUTBF, bool KAL, bool NAL, bool TDM>
__global__ __launch_bounds__(256)
void gemm_wmma(const void* __restrict__ Aptr, const float* __restrict__ Bw,
               const float* __restrict__ bias, void* __restrict__ Cptr,
               int M, int N, int Kd, int lda, int ldb, int ldc)
{
  __shared__ bf16_t As[BM * LDS_S];
  __shared__ bf16_t Bs[BN * LDS_S];
  const int tid  = threadIdx.x;
  const int lane = tid & 31;
  const int w    = tid >> 5;
  const int wm   = w & 3;     // 0..3 -> M offset wm*32
  const int wn   = w >> 2;    // 0..1 -> N offset wn*32
  const int m0   = blockIdx.y * BM;
  const int n0   = blockIdx.x * BN;

  f32x8 acc[2][2];
  #pragma unroll
  for (int i = 0; i < 2; i++)
    #pragma unroll
    for (int j = 0; j < 2; j++)
      #pragma unroll
      for (int e = 0; e < 8; e++) acc[i][j][e] = 0.0f;

  const int arow  = tid >> 1;          // 0..127
  const int akoff = (tid & 1) * 16;    // 0 / 16
  const int brow  = tid >> 2;          // 0..63
  const int bkoff = (tid & 3) * 8;     // 0,8,16,24
  const int grp   = lane >> 4;
  const int lr    = lane & 15;

  for (int kt = 0; kt < Kd; kt += BK) {
    // ---- stage A tile ----
    if constexpr (TDM) {
      // one wave issues the DMA for the whole block (TDM ignores EXEC)
      if (tid < 32) {
        const bf16_t* src = (const bf16_t*)Aptr + (size_t)m0 * lda + kt;
        tdm_load_a_bf16(As, src, Kd - kt, M - m0, lda);
      }
    } else if constexpr (ABF) {
      const int gm = m0 + arow;
      bf16_t* dst = &As[arow * LDS_S + akoff];
      const bf16_t* src = (const bf16_t*)Aptr + (size_t)gm * lda + kt + akoff;
      if (KAL || (kt + akoff + 15 < Kd)) {
        uint4 v0 = *(const uint4*)(src);
        uint4 v1 = *(const uint4*)(src + 8);
        *(uint4*)(dst)     = v0;
        *(uint4*)(dst + 8) = v1;
      } else {
        #pragma unroll
        for (int i = 0; i < 16; i++)
          dst[i] = (kt + akoff + i < Kd) ? src[i] : (bf16_t)0.0f;
      }
    } else {
      const int gm = m0 + arow;
      bf16_t* dst = &As[arow * LDS_S + akoff];
      const float* src = (const float*)Aptr + (size_t)gm * lda + kt + akoff;
      if (KAL || (kt + akoff + 15 < Kd)) {
        #pragma unroll
        for (int qd = 0; qd < 4; qd++) {
          float4 v = *(const float4*)(src + qd * 4);
          dst[qd*4+0] = (bf16_t)v.x; dst[qd*4+1] = (bf16_t)v.y;
          dst[qd*4+2] = (bf16_t)v.z; dst[qd*4+3] = (bf16_t)v.w;
        }
      } else {
        #pragma unroll
        for (int i = 0; i < 16; i++)
          dst[i] = (kt + akoff + i < Kd) ? (bf16_t)src[i] : (bf16_t)0.0f;
      }
    }
    // ---- stage B tile (weights, f32 -> bf16) ----
    {
      const int gn = n0 + brow;
      bf16_t* dst = &Bs[brow * LDS_S + bkoff];
      if (NAL || gn < N) {
        const float* src = Bw + (size_t)gn * ldb + kt + bkoff;
        if (KAL || (kt + bkoff + 7 < Kd)) {
          #pragma unroll
          for (int qd = 0; qd < 2; qd++) {
            float4 v = *(const float4*)(src + qd * 4);
            dst[qd*4+0] = (bf16_t)v.x; dst[qd*4+1] = (bf16_t)v.y;
            dst[qd*4+2] = (bf16_t)v.z; dst[qd*4+3] = (bf16_t)v.w;
          }
        } else {
          #pragma unroll
          for (int i = 0; i < 8; i++)
            dst[i] = (kt + bkoff + i < Kd) ? (bf16_t)src[i] : (bf16_t)0.0f;
        }
      } else {
        #pragma unroll
        for (int i = 0; i < 8; i++) dst[i] = (bf16_t)0.0f;
      }
    }
    if constexpr (TDM) {
      if (tid < 32) __builtin_amdgcn_s_wait_tensorcnt(0);
    }
    __syncthreads();

    // ---- fragments + 4 WMMA ----
    bf16x16 afrag[2], bfrag[2];
    #pragma unroll
    for (int mi = 0; mi < 2; mi++) {
      // A layout: elem e -> K = (e<8?0:16) + grp*8 + (e&7)
      const bf16_t* ap = &As[(wm * 32 + mi * 16 + lr) * LDS_S + grp * 8];
      bf16x8 lo = *(const bf16x8*)ap;
      bf16x8 hi = *(const bf16x8*)(ap + 16);
      #pragma unroll
      for (int e = 0; e < 8; e++) { afrag[mi][e] = lo[e]; afrag[mi][8 + e] = hi[e]; }
    }
    #pragma unroll
    for (int ni = 0; ni < 2; ni++) {
      // B layout: elem e -> K = (e<8?0:16) + grp*8 + (e&7) for column n = lane&15
      const bf16_t* bp = &Bs[(wn * 32 + ni * 16 + lr) * LDS_S + grp * 8];
      bf16x8 lo = *(const bf16x8*)bp;
      bf16x8 hi = *(const bf16x8*)(bp + 16);
      #pragma unroll
      for (int e = 0; e < 8; e++) { bfrag[ni][e] = lo[e]; bfrag[ni][8 + e] = hi[e]; }
    }
    #pragma unroll
    for (int mi = 0; mi < 2; mi++)
      #pragma unroll
      for (int ni = 0; ni < 2; ni++)
        acc[mi][ni] = __builtin_amdgcn_wmma_f32_16x16x32_bf16(
            false, afrag[mi], false, bfrag[ni], (short)0, acc[mi][ni], false, false);
    __syncthreads();
  }

  // ---- epilogue + store. C layout: VGPR v -> M = v + (lane>=16?8:0), N = lane&15
  const int moff = grp * 8;
  #pragma unroll
  for (int mi = 0; mi < 2; mi++) {
    #pragma unroll
    for (int ni = 0; ni < 2; ni++) {
      int gn = n0 + wn * 32 + ni * 16 + lr;
      if (!NAL && gn >= N) continue;
      float bv = (EPI != EPI_NONE) ? bias[gn] : 0.0f;
      #pragma unroll
      for (int v = 0; v < 8; v++) {
        int gm = m0 + wm * 32 + mi * 16 + moff + v;
        float val = acc[mi][ni][v];
        if constexpr (EPI == EPI_BIAS) {
          val += bv;
        } else if constexpr (EPI == EPI_BIAS_GELU) {
          val += bv;
          float c = val;
          val = 0.5f * c * (1.0f + tanhf(0.7978845608f * (c + 0.044715f * c * c * c)));
        } else if constexpr (EPI == EPI_BIAS_SOFTPLUS) {
          val += bv;
          val = (val > 20.0f) ? val : log1pf(__expf(val));
        }
        if constexpr (OUTBF) ((bf16_t*)Cptr)[(size_t)gm * ldc + gn] = (bf16_t)val;
        else                 ((float*)Cptr)[(size_t)gm * ldc + gn]  = val;
      }
    }
  }
}

// ---------------- gathered causal depthwise conv (K=4) + SiLU, 4 paths ----
__global__ __launch_bounds__(256)
void k_conv4(const float* __restrict__ xz, const int* __restrict__ zz,
             const float* __restrict__ cw, const float* __restrict__ cb,
             float* __restrict__ xconv4)
{
  const int r  = blockIdx.x;           // p*B*L + b*L + j
  const int j  = r & (LL - 1);
  const int pb = r >> 10;
  const int b  = pb & (BB - 1);
  const int p  = pb >> 4;
  int ti[KK];
  #pragma unroll
  for (int k = 0; k < KK; k++) {
    int i = j - (KK - 1) + k;
    ti[k] = (i >= 0) ? zz[p * LL + i] : -1;
  }
  float* out = xconv4 + (size_t)r * DII;
  #pragma unroll
  for (int c = 0; c < 3; c++) {
    int d = threadIdx.x + c * 256;
    float s = cb[d];
    #pragma unroll
    for (int k = 0; k < KK; k++)
      if (ti[k] >= 0)
        s += cw[d * KK + k] * xz[((size_t)b * LL + ti[k]) * (2 * DII) + d];
    s = s / (1.0f + __expf(-s));   // silu
    out[d] = s;
  }
}

// ---------------- selective scan, parallel over (path, batch, channel) ----
__global__ __launch_bounds__(256)
void k_scan(const float* __restrict__ dt4, const float* __restrict__ proj4,
            const float* __restrict__ xconv4, const float* __restrict__ xz,
            const int* __restrict__ zz, const float* __restrict__ A_log,
            const float* __restrict__ Dp, float* __restrict__ yg4)
{
  const int pb = blockIdx.x;          // p*B + b
  const int b  = pb & (BB - 1);
  const int p  = pb >> 4;
  const int d  = blockIdx.y * 256 + threadIdx.x;
  float Ad[DSZ];
  #pragma unroll
  for (int n = 0; n < DSZ; n++) Ad[n] = -__expf(A_log[d * DSZ + n]);
  const float Dd = Dp[d];
  float h[DSZ];
  #pragma unroll
  for (int n = 0; n < DSZ; n++) h[n] = 0.f;
  __shared__ float sBC[2 * DSZ];
  const size_t rb = (size_t)pb * LL;
  for (int j = 0; j < LL; j++) {
    const size_t base = rb + j;
    if (threadIdx.x < 2 * DSZ)
      sBC[threadIdx.x] = proj4[base * (DRR + 2 * DSZ) + DRR + threadIdx.x];
    __syncthreads();
    float dt  = dt4[base * DII + d];
    float xv  = xconv4[base * DII + d];
    float dtx = dt * xv;
    float y = 0.f;
    #pragma unroll
    for (int n = 0; n < DSZ; n++) {
      h[n] = __expf(dt * Ad[n]) * h[n] + dtx * sBC[n];
      y += h[n] * sBC[DSZ + n];
    }
    int tok = zz[p * LL + j];
    float z = xz[((size_t)b * LL + tok) * (2 * DII) + DII + d];
    float yg = (y + Dd * xv) * (z / (1.0f + __expf(-z)));
    yg4[base * DII + d] = yg;
    __syncthreads();
  }
}

// ---------------- gather-reduce 4 paths back to original order ------------
__global__ __launch_bounds__(256)
void k_reduce(const float* __restrict__ yg4, const int* __restrict__ zzrev,
              float* __restrict__ acc)
{
  const int token = blockIdx.x;
  const int b = token >> 10;
  const int l = token & (LL - 1);
  #pragma unroll
  for (int c = 0; c < 3; c++) {
    int d = threadIdx.x + c * 256;
    float s = 0.f;
    #pragma unroll
    for (int p = 0; p < NPATH; p++) {
      int jp = zzrev[p * LL + l];
      s += yg4[(((size_t)p * BB + b) * LL + jp) * DII + d];
    }
    acc[(size_t)token * DII + d] = 0.25f * s;
  }
}

// ---------------- final residual: out += g_mlp * h2 ------------------------
__global__ __launch_bounds__(256)
void k_final(float* __restrict__ out, const float* __restrict__ t,
             const float* __restrict__ h2)
{
  size_t idx = (size_t)blockIdx.x * 256 + threadIdx.x;
  int d = (int)(idx % HH);
  int b = (int)(idx / ((size_t)LL * HH));
  out[idx] += t[(size_t)b * 6 * HH + 5 * HH + d] * h2[idx];
}

// ===========================================================================
extern "C" void kernel_launch(void* const* d_in, const int* in_sizes, int n_in,
                              void* d_out, int out_size, void* d_ws, size_t ws_size,
                              hipStream_t stream)
{
  (void)in_sizes; (void)n_in; (void)out_size; (void)ws_size;
  const float* x    = (const float*)d_in[0];
  const float* t    = (const float*)d_in[1];
  const float* y    = (const float*)d_in[2];
  const int*   zz   = (const int*)d_in[3];
  const int*   zzr  = (const int*)d_in[4];
  const float* Win  = (const float*)d_in[5];   // [2*DI, H]
  const float* cw   = (const float*)d_in[6];   // [DI, K]
  const float* cb   = (const float*)d_in[7];
  const float* Wxp  = (const float*)d_in[8];   // [80, DI]
  const float* Wdt  = (const float*)d_in[9];   // [DI, DR]
  const float* bdt  = (const float*)d_in[10];
  const float* Alog = (const float*)d_in[11];  // [DI, DS]
  const float* Dp   = (const float*)d_in[12];
  const float* Wout = (const float*)d_in[13];  // [H, DI]
  const float* W1   = (const float*)d_in[14];  // [MLPH, H]
  const float* b1   = (const float*)d_in[15];
  const float* W2   = (const float*)d_in[16];  // [H, MLPH]
  const float* b2   = (const float*)d_in[17];
  float* out = (float*)d_out;

  const int M  = BB * LL;        // 16384
  const int M4 = NPATH * M;      // 65536

  char* ws = (char*)d_ws;
  size_t off = 0;
  auto alloc = [&](size_t bytes) -> char* {
    char* p = ws + off;
    off += (bytes + 255) & ~(size_t)255;
    return p;
  };
  bf16_t* tmpbf  = (bf16_t*)alloc((size_t)M  * HH   * 2);
  float*  xzbuf  = (float*) alloc((size_t)M  * 2*DII* 4);
  float*  xconv4 = (float*) alloc((size_t)M4 * DII  * 4);
  float*  proj4  = (float*) alloc((size_t)M4 * 80   * 4);
  float*  dt4    = (float*) alloc((size_t)M4 * DII  * 4);
  float*  yg4    = (float*) alloc((size_t)M4 * DII  * 4);
  float*  accb   = (float*) alloc((size_t)M  * DII  * 4);
  // reuse (stream-ordered, producers finish before overwrite):
  float*  mam  = dt4;                 // free after scan
  bf16_t* hmod = tmpbf;               // free after in_proj GEMM
  bf16_t* h1g  = (bf16_t*)xconv4;     // free after scan
  float*  h2   = yg4;                 // free after reduce

  dim3 blk(256);

  // 1. modulated LN -> bf16
  k_ln_mod<0><<<M, blk, 0, stream>>>(x, t, nullptr, nullptr, nullptr, tmpbf, 0, 1, 0);
  // 2. in_proj (once for all 4 paths): xz = tmp @ Win^T   [M x 1536], TDM A-feed
  gemm_wmma<EPI_NONE, true, false, true, true, true>
      <<<dim3((2*DII)/BN, M/BM), blk, 0, stream>>>(
      tmpbf, Win, nullptr, xzbuf, M, 2*DII, HH, HH, HH, 2*DII);
  // 3. gathered conv + silu for all 4 paths
  k_conv4<<<M4, blk, 0, stream>>>(xzbuf, zz, cw, cb, xconv4);
  // 4. x_proj over all 4 paths batched: [M4 x 80]  (K aligned, N ragged)
  gemm_wmma<EPI_NONE, false, false, true, false, false>
      <<<dim3((80 + BN - 1)/BN, M4/BM), blk, 0, stream>>>(
      xconv4, Wxp, nullptr, proj4, M4, 80, DII, DII, DII, 80);
  // 5. dt_proj + softplus: [M4 x 768], K=48 (ragged, zero-padded), A stride 80
  gemm_wmma<EPI_BIAS_SOFTPLUS, false, false, false, true, false>
      <<<dim3(DII/BN, M4/BM), blk, 0, stream>>>(
      proj4, Wdt, bdt, dt4, M4, DII, DRR, 80, DRR, DII);
  // 6. selective scan (all paths / batches / channel chunks in parallel)
  k_scan<<<dim3(NPATH*BB, DII/256), blk, 0, stream>>>(
      dt4, proj4, xconv4, xzbuf, zz, Alog, Dp, yg4);
  // 7. gather-reduce 4 paths -> acc (original order, *0.25)
  k_reduce<<<M, blk, 0, stream>>>(yg4, zzr, accb);
  // 8. out_proj (once): mam = acc @ Wout^T   [M x 768]
  gemm_wmma<EPI_NONE, false, false, true, true, false>
      <<<dim3(HH/BN, M/BM), blk, 0, stream>>>(
      accb, Wout, nullptr, mam, M, HH, DII, DII, DII, HH);
  // 9. x1 = x + g_msa*mam + y  -> d_out ; then LN+mod -> hmod (bf16)
  k_ln_mod<1><<<M, blk, 0, stream>>>(x, t, mam, y, out, hmod, 3, 4, 2);
  // 10. MLP fc1 + bias + gelu -> bf16  [M x 3072], TDM A-feed
  gemm_wmma<EPI_BIAS_GELU, true, true, true, true, true>
      <<<dim3(MLPH/BN, M/BM), blk, 0, stream>>>(
      hmod, W1, b1, h1g, M, MLPH, HH, HH, HH, MLPH);
  // 11. MLP fc2 + bias  [M x 768], TDM A-feed
  gemm_wmma<EPI_BIAS, true, false, true, true, true>
      <<<dim3(HH/BN, M/BM), blk, 0, stream>>>(
      h1g, W2, b2, h2, M, HH, MLPH, MLPH, MLPH, HH);
  // 12. out = x1 + g_mlp * h2
  k_final<<<((size_t)M * HH) / 256, blk, 0, stream>>>(out, t, h2);
}
